// QKVAttention_53386443489866
// MI455X (gfx1250) — matrix operand order repaired and checked by reading, not compiled, not run on hardware
//
#include <hip/hip_runtime.h>

typedef __attribute__((ext_vector_type(16))) __bf16 v16bf;
typedef __attribute__((ext_vector_type(8)))  __bf16 v8bf;
typedef __attribute__((ext_vector_type(4)))  __bf16 v4bf;
typedef __attribute__((ext_vector_type(8)))  float  v8f;

#define N_BATCH 64
#define C_DIM   64
#define T_DIM   2048
#define TILE_T  128             // query rows per workgroup
#define TQ      16              // query rows per wave
#define SCH     64              // key/value chunk per iteration
#define WAVES   8
#define NIT     (T_DIM / SCH)   // 32 iterations

// ---- DPP16 16-lane butterfly reduction (VALU-only, co-executes with WMMA) ----
template<int CTRL>
__device__ __forceinline__ float dpp_mov(float v) {
    return __int_as_float(__builtin_amdgcn_update_dpp(
        0, __float_as_int(v), CTRL, 0xF, 0xF, true));
}
__device__ __forceinline__ float rowmax16(float v) {
    v = fmaxf(v, dpp_mov<0xB1>(v));    // quad_perm [1,0,3,2]  (xor 1)
    v = fmaxf(v, dpp_mov<0x4E>(v));    // quad_perm [2,3,0,1]  (xor 2)
    v = fmaxf(v, dpp_mov<0x141>(v));   // row_half_mirror      (combine quads)
    v = fmaxf(v, dpp_mov<0x140>(v));   // row_mirror           (combine octets)
    return v;
}
__device__ __forceinline__ float rowsum16(float v) {
    v += dpp_mov<0xB1>(v);
    v += dpp_mov<0x4E>(v);
    v += dpp_mov<0x141>(v);
    v += dpp_mov<0x140>(v);
    return v;
}

__global__ __launch_bounds__(256) void qkv_attn_kernel(const float* __restrict__ qkv,
                                                       float* __restrict__ out)
{
    // Ping-pong K/V staging + per-wave P relayout tiles. 48 KB total.
    __shared__ alignas(32) __bf16 kT[2][SCH][C_DIM];     // [buf][s][c]  16 KB
    __shared__ alignas(32) __bf16 vS[2][C_DIM][SCH];     // [buf][c][s]  16 KB
    __shared__ alignas(32) __bf16 pP[WAVES][TQ][SCH];    //              16 KB

    const int b     = blockIdx.x / (T_DIM / TILE_T);
    const int tTile = blockIdx.x % (T_DIM / TILE_T);
    const int tid   = threadIdx.x;
    const int lane  = tid & 31;
    const int wave  = tid >> 5;
    const int ln16  = lane & 15;
    const int hh    = lane >> 4;

    const size_t baseQ = (size_t)b * (3 * C_DIM) * T_DIM;
    const float* qp = qkv + baseQ;
    const float* kp = qkv + baseQ + (size_t)C_DIM * T_DIM;
    const float* vp = qkv + baseQ + (size_t)(2 * C_DIM) * T_DIM;

    const int qBase = tTile * TILE_T + wave * TQ;

    // ---- Q A-fragments: 16(M=t) x 64(K=c) split into 2 K-chunks of 32.
    // Combined softmax scale 1/sqrt(C)=1/8 folded into Q.
    v16bf aq[2];
    {
        const int tq = qBase + ln16;
        #pragma unroll
        for (int h2 = 0; h2 < 2; ++h2)
            #pragma unroll
            for (int i = 0; i < 16; ++i) {
                const int c = 32 * h2 + 16 * hh + i;
                aq[h2][i] = (__bf16)(qp[(size_t)c * T_DIM + tq] * 0.125f);
            }
    }

    const v8f zero8 = {0.f, 0.f, 0.f, 0.f, 0.f, 0.f, 0.f, 0.f};
    v8f acc[4] = {zero8, zero8, zero8, zero8};
    float mrow[8], lrow[8];
    #pragma unroll
    for (int r = 0; r < 8; ++r) { mrow[r] = -1e30f; lrow[r] = 0.f; }

    // ---- software pipeline helpers: 4 float4 per tensor per thread per chunk
    float4 kr[4], vr[4];
    auto loadregs = [&](int s0) {
        #pragma unroll
        for (int rr = 0; rr < 4; ++rr) {
            const int idx = tid + rr * 256;      // 64 c-rows x 16 float4 groups
            const int c   = idx >> 4;
            const int s4  = (idx & 15) * 4;
            kr[rr] = *(const float4*)(kp + (size_t)c * T_DIM + s0 + s4);
            vr[rr] = *(const float4*)(vp + (size_t)c * T_DIM + s0 + s4);
        }
    };
    auto stageLDS = [&](int buf) {
        #pragma unroll
        for (int rr = 0; rr < 4; ++rr) {
            const int idx = tid + rr * 256;
            const int c   = idx >> 4;
            const int s4  = (idx & 15) * 4;
            kT[buf][s4 + 0][c] = (__bf16)kr[rr].x;   // transposed scatter
            kT[buf][s4 + 1][c] = (__bf16)kr[rr].y;
            kT[buf][s4 + 2][c] = (__bf16)kr[rr].z;
            kT[buf][s4 + 3][c] = (__bf16)kr[rr].w;
            v4bf pv;                                  // packed 8B store
            pv[0] = (__bf16)vr[rr].x; pv[1] = (__bf16)vr[rr].y;
            pv[2] = (__bf16)vr[rr].z; pv[3] = (__bf16)vr[rr].w;
            *(v4bf*)&vS[buf][c][s4] = pv;
        }
    };

    loadregs(0);
    stageLDS(0);
    __syncthreads();

    for (int it = 0; it < NIT; ++it) {
        const int cur = it & 1;
        const bool more = (it + 1 < NIT);
        if (more) loadregs((it + 1) * SCH);   // issue next chunk early

        // ---- S = Q K^T : 16(t) x 64(s) as four 16x16 N-tiles, K(=c)=64 -> 2 WMMAs each
        v8f st[4];
        #pragma unroll
        for (int n16 = 0; n16 < 4; ++n16) {
            const int sl = n16 * 16 + ln16;
            v8f c0 = zero8;
            #pragma unroll
            for (int h2 = 0; h2 < 2; ++h2) {
                const v16bf bk = *(const v16bf*)&kT[cur][sl][32 * h2 + 16 * hh];
                c0 = __builtin_amdgcn_wmma_f32_16x16x32_bf16(false, aq[h2], false, bk,
                                                             (short)0, c0, false, false);
            }
            st[n16] = c0;
        }

        // ---- online softmax over 64 keys (rows M = r + 8*hh; DPP butterflies)
        #pragma unroll
        for (int r = 0; r < 8; ++r) {
            float mx = fmaxf(fmaxf(st[0][r], st[1][r]), fmaxf(st[2][r], st[3][r]));
            mx = rowmax16(mx);
            const float mnew = fmaxf(mrow[r], mx);
            const float corr = __expf(mrow[r] - mnew);
            float p[4];
            float sm = 0.f;
            #pragma unroll
            for (int n = 0; n < 4; ++n) { p[n] = __expf(st[n][r] - mnew); sm += p[n]; }
            sm = rowsum16(sm);
            mrow[r] = mnew;
            lrow[r] = lrow[r] * corr + sm;
            #pragma unroll
            for (int k = 0; k < 4; ++k) acc[k][r] *= corr;
            #pragma unroll
            for (int n = 0; n < 4; ++n)
                pP[wave][r + 8 * hh][n * 16 + ln16] = (__bf16)p[n];
        }

        // ---- reload P as A-fragments 16(t) x 64(K=s): 2 chunks of 32
        v16bf ap[2];
        #pragma unroll
        for (int h2 = 0; h2 < 2; ++h2) {
            const v8bf plo = *(const v8bf*)&pP[wave][ln16][32 * h2 + 16 * hh];
            const v8bf phi = *(const v8bf*)&pP[wave][ln16][32 * h2 + 16 * hh + 8];
            ap[h2] = __builtin_shufflevector(plo, phi,
                       0,1,2,3,4,5,6,7,8,9,10,11,12,13,14,15);
        }

        // ---- acc += P * V : 4 c-chunks x 2 s-chunks
        #pragma unroll
        for (int k = 0; k < 4; ++k)
            #pragma unroll
            for (int h2 = 0; h2 < 2; ++h2) {
                const v16bf bv = *(const v16bf*)&vS[cur][16 * k + ln16][32 * h2 + 16 * hh];
                acc[k] = __builtin_amdgcn_wmma_f32_16x16x32_bf16(false, ap[h2], false, bv,
                                                                 (short)0, acc[k], false, false);
            }

        if (more) stageLDS(cur ^ 1);   // write next buffer (safe: one barrier/iter)
        __syncthreads();
    }

    // ---- epilogue: out[b, c, t] = acc / l
    float* op = out + (size_t)b * C_DIM * T_DIM;
    #pragma unroll
    for (int r = 0; r < 8; ++r) {
        const float inv = 1.0f / lrow[r];
        const int t = qBase + r + 8 * hh;
        #pragma unroll
        for (int k = 0; k < 4; ++k) {
            const int c = 16 * k + ln16;
            op[(size_t)c * T_DIM + t] = acc[k][r] * inv;
        }
    }
}

extern "C" void kernel_launch(void* const* d_in, const int* in_sizes, int n_in,
                              void* d_out, int out_size, void* d_ws, size_t ws_size,
                              hipStream_t stream) {
    (void)in_sizes; (void)n_in; (void)out_size; (void)d_ws; (void)ws_size;
    const float* qkv = (const float*)d_in[0];
    float* out = (float*)d_out;
    dim3 grid(N_BATCH * (T_DIM / TILE_T));   // 64 batches x 16 query tiles
    dim3 block(256);                          // 8 wave32 waves
    hipLaunchKernelGGL(qkv_attn_kernel, grid, block, 0, stream, qkv, out);
}
